// CorticalGNN_37692632990313
// MI455X (gfx1250) — matrix operand order repaired and checked by reading, not compiled, not run on hardware
//
#include <hip/hip_runtime.h>

#define NN 8192
#define DD 512

typedef __attribute__((ext_vector_type(16))) __bf16        v16bf;
typedef __attribute__((ext_vector_type(2)))  __bf16        v2bf;
typedef __attribute__((ext_vector_type(8)))  float         v8f;
typedef __attribute__((ext_vector_type(4)))  unsigned int  u32x4;
typedef __attribute__((ext_vector_type(4)))  int           i32x4;
typedef __attribute__((ext_vector_type(4)))  float         f32x4;

struct U32x8 { u32x4 lo, hi; };
static_assert(sizeof(U32x8) == 32, "pack size");
static_assert(sizeof(v16bf) == 32, "frag size");

#define HAS_ASYNC_LDS __has_builtin(__builtin_amdgcn_global_load_async_to_lds_b128)

static __device__ __forceinline__ void wait_asynccnt0() {
#if __has_builtin(__builtin_amdgcn_s_wait_asynccnt)
  __builtin_amdgcn_s_wait_asynccnt(0);
#else
  asm volatile("s_wait_asynccnt 0x0" ::: "memory");
#endif
}

static __device__ __forceinline__ unsigned short f2bf(float f) {
  unsigned u = __builtin_bit_cast(unsigned int, f);
  u += 0x7FFFu + ((u >> 16) & 1u);        // round-to-nearest-even
  return (unsigned short)(u >> 16);
}
static __device__ __forceinline__ unsigned pk2(float lo, float hi) {
#if __has_builtin(__builtin_amdgcn_cvt_pk_bf16_f32)
  v2bf r = __builtin_amdgcn_cvt_pk_bf16_f32(lo, hi);
  return __builtin_bit_cast(unsigned int, r);
#else
  return (unsigned)f2bf(lo) | ((unsigned)f2bf(hi) << 16);
#endif
}
static __device__ __forceinline__ v16bf frag_from(u32x4 a, u32x4 b) {
  U32x8 t{a, b};
  return __builtin_bit_cast(v16bf, t);
}

// ---------------------------------------------------------------------------
// Prep: xT[n][k] = bf16(x[k][n])   (32x32 LDS-tiled transpose, packed stores)
// ---------------------------------------------------------------------------
__global__ __launch_bounds__(256) void prep_xT(const float* __restrict__ x,
                                               unsigned short* __restrict__ xT) {
  __shared__ unsigned short tile[32][33];
  const int k0 = blockIdx.x * 32;   // over 8192 -> gridDim.x = 256
  const int n0 = blockIdx.y * 32;   // over 512  -> gridDim.y = 16
  const int tx = threadIdx.x;       // 32
  const int ty = threadIdx.y;       // 8
#pragma unroll
  for (int i = 0; i < 32; i += 8)   // tile[n_local][k_local]
    tile[tx][ty + i] = f2bf(x[(size_t)(k0 + ty + i) * DD + n0 + tx]);
  __syncthreads();
  const int t    = ty * 32 + tx;    // 0..255
  const int nrow = t >> 3;          // 0..31
  const int kc   = t & 7;           // 0..7
#pragma unroll
  for (int p = 0; p < 2; ++p) {
    int ku = kc + p * 8;            // uint column 0..15 (2 bf16 each)
    unsigned v = (unsigned)tile[nrow][2 * ku] |
                 ((unsigned)tile[nrow][2 * ku + 1] << 16);
    *(unsigned*)(xT + (size_t)(n0 + nrow) * NN + k0 + 2 * ku) = v;
  }
}

// Prep: Wb = bf16(W), row-major [out][in] (== B-friendly layout for agg @ W^T)
__global__ __launch_bounds__(256) void prep_W(const float* __restrict__ W,
                                              unsigned short* __restrict__ Wb) {
  size_t id = (size_t)blockIdx.x * 256 + threadIdx.x;   // 131072 threads, 2 elems each
  const float* p = W + 2 * id;
  *(unsigned*)(Wb + 2 * id) = pk2(p[0], p[1]);
}

// ---------------------------------------------------------------------------
// Stage-1 staging helpers
// ---------------------------------------------------------------------------
// B chunk: 32 KB of bf16 xT rows -> LDS. Async (no VGPR staging) if available.
static __device__ __forceinline__ void b_fetch_async(
    const unsigned short* __restrict__ xT, unsigned short* bB, int k0, int t) {
#if HAS_ASYNC_LDS
#pragma unroll
  for (int rr = 0; rr < 2; ++rr) {
    int n = t + rr * 256;
    const unsigned short* g = xT + (size_t)n * NN + k0;
    unsigned short* l = bB + n * 40;
#pragma unroll
    for (int i = 0; i < 4; ++i) {
      __builtin_amdgcn_global_load_async_to_lds_b128(
          (__attribute__((address_space(1))) i32x4*)(g + i * 8),
          (__attribute__((address_space(3))) i32x4*)(l + i * 8), 0, 0);
    }
  }
#else
#pragma unroll
  for (int rr = 0; rr < 2; ++rr) {
    int n = t + rr * 256;
    const u32x4* src = (const u32x4*)(xT + (size_t)n * NN + k0);
    u32x4 v0 = src[0], v1 = src[1], v2 = src[2], v3 = src[3];
    u32x4* dst = (u32x4*)(bB + n * 40);
    dst[0] = v0; dst[1] = v1; dst[2] = v2; dst[3] = v3;
  }
#endif
}

// A chunk: 8 adj floats per thread (stream-once, NT), converted on store.
struct AStage { f32x4 a0, a1; };

static __device__ __forceinline__ void a_fetch(AStage& s,
    const float* __restrict__ aRow, int k0) {
  const f32x4* sp = (const f32x4*)(aRow + k0);   // aRow pre-offset by q*8
  s.a0 = __builtin_nontemporal_load(sp);
  s.a1 = __builtin_nontemporal_load(sp + 1);
}
static __device__ __forceinline__ void a_store(const AStage& s,
    unsigned short* aB, int t, int q) {
  u32x4 v{pk2(s.a0.x, s.a0.y), pk2(s.a0.z, s.a0.w),
          pk2(s.a1.x, s.a1.y), pk2(s.a1.z, s.a1.w)};
  *(u32x4*)(aB + (t >> 2) * 40 + q * 8) = v;
}

// ---------------------------------------------------------------------------
// Fused: agg = adj @ x  (bf16 WMMA, K=8192), then out = relu(agg @ W^T + b)
// Block = 64-row panel, 256 threads = 8 waves = 4 M-tiles x 2 N-halves.
// Double-buffered LDS pipeline, async B staging: one barrier per K chunk.
// Inner 16-tile loop manually software-pipelined (B fragment loaded 1 ahead).
// ---------------------------------------------------------------------------
#define BUFBYTES 46080   // B [512][40] (40960B) + A [64][40] (5120B)

__global__ __launch_bounds__(256, 1) void gnn_fused(
    const float* __restrict__ adj, const unsigned short* __restrict__ xT,
    const unsigned short* __restrict__ Wb, const float* __restrict__ bias,
    float* __restrict__ out) {
  __shared__ __align__(16) unsigned char smem[2 * BUFBYTES];  // 92160 B
  unsigned short* aggL = (unsigned short*)smem;               // stage2 A: [64][520]

  const int t    = threadIdx.x;
  const int lane = t & 31;
  const int wave = t >> 5;
  const int mt   = wave & 3;    // M-tile within panel
  const int nh   = wave >> 2;   // N-half (0: cols 0..255, 1: cols 256..511)
  const int ml   = lane & 15;   // row within tile (A) / col within tile (B)
  const int hf   = lane >> 4;   // lane half -> K sub-run select
  const int row0 = blockIdx.x * 64;

  v8f zero = {0.f, 0.f, 0.f, 0.f, 0.f, 0.f, 0.f, 0.f};
  v8f acc[16];
#pragma unroll
  for (int i = 0; i < 16; ++i) acc[i] = zero;

  const int q = t & 3;  // quarter of the 32-wide K chunk this thread stages
  const float* aRow = adj + (size_t)(row0 + (t >> 2)) * NN + q * 8;

  // ---------------- Stage 1: agg = adj @ x ----------------
  AStage ast;
  b_fetch_async(xT, (unsigned short*)smem, 0, t);
  a_fetch(ast, aRow, 0);
  a_store(ast, (unsigned short*)(smem + 40960), t, q);
  wait_asynccnt0();
  __syncthreads();

  for (int k0 = 0; k0 < NN; k0 += 32) {
    const int cb = (k0 >> 5) & 1;
    const unsigned short* curB = (const unsigned short*)(smem + cb * BUFBYTES);
    const unsigned short* curA = curB + 20480;   // +40960 bytes
    unsigned short* nxtB = (unsigned short*)(smem + (cb ^ 1) * BUFBYTES);
    const bool more = (k0 + 32) < NN;

    if (more) {
      b_fetch_async(xT, nxtB, k0 + 32, t);  // async copy overlaps the WMMAs below
      a_fetch(ast, aRow, k0 + 32);
    }

    // A fragment (ISA 16-bit A layout: per lane-half, runs [8h,8h+8) and [16+8h,+8))
    const unsigned short* ab = curA + (mt * 16 + ml) * 40;
    v16bf afrag = frag_from(*(const u32x4*)(ab + 8 * hf),
                            *(const u32x4*)(ab + 16 + 8 * hf));

    // B fragments: column n = nh*256 + nt*16 + ml, K runs [8h,8h+8) and [16+8h,+8).
    const unsigned short* bbase = curB + (nh * 256 + ml) * 40 + 8 * hf;
    auto ldb = [&](int nt) -> v16bf {
      const unsigned short* bb = bbase + nt * (16 * 40);
      return frag_from(*(const u32x4*)bb, *(const u32x4*)(bb + 16));
    };
    v16bf bf = ldb(0);
#pragma unroll
    for (int nt = 0; nt < 16; ++nt) {
      v16bf bn = bf;
      if (nt < 15) bn = ldb(nt + 1);   // issue next tile's loads before this WMMA
      acc[nt] = __builtin_amdgcn_wmma_f32_16x16x32_bf16(
          false, afrag, false, bf, (short)0, acc[nt], false, false);
      bf = bn;
    }

    if (more) a_store(ast, nxtB + 20480, t, q);
    wait_asynccnt0();   // async B copy for next chunk landed in LDS
    __syncthreads();    // visible to all waves; this chunk's reads complete
  }

  // ---------------- agg panel -> LDS as bf16 (row stride 520) ----------------
#pragma unroll
  for (int nt = 0; nt < 16; ++nt) {
#pragma unroll
    for (int r = 0; r < 8; ++r) {
      // C layout: M = r + 8*hf, N = ml (+ tile bases)
      aggL[(mt * 16 + 8 * hf + r) * 520 + nh * 256 + nt * 16 + ml] = f2bf(acc[nt][r]);
    }
  }
  __syncthreads();

  // ---------------- Stage 2: out = relu(agg @ W^T + b) ----------------
  v8f acc2[16];
#pragma unroll
  for (int i = 0; i < 16; ++i) acc2[i] = zero;

  for (int k0 = 0; k0 < DD; k0 += 32) {
    const unsigned short* ab = aggL + (mt * 16 + ml) * 520 + k0;
    v16bf afrag = frag_from(*(const u32x4*)(ab + 8 * hf),
                            *(const u32x4*)(ab + 16 + 8 * hf));
    // B = W^T fragment: row o = output col, contiguous i = K  (Wb row-major)
    const unsigned short* wbase = Wb + (size_t)(nh * 256 + ml) * DD + k0 + 8 * hf;
    auto ldw = [&](int nt) -> v16bf {
      const unsigned short* wb = wbase + nt * (16 * DD);
      return frag_from(*(const u32x4*)wb, *(const u32x4*)(wb + 16));
    };
    v16bf bf = ldw(0);
#pragma unroll
    for (int nt = 0; nt < 16; ++nt) {
      v16bf bn = bf;
      if (nt < 15) bn = ldw(nt + 1);
      acc2[nt] = __builtin_amdgcn_wmma_f32_16x16x32_bf16(
          false, afrag, false, bf, (short)0, acc2[nt], false, false);
      bf = bn;
    }
  }

  // epilogue: bias + ReLU, stream-out (non-temporal)
  const int mg = row0 + mt * 16 + 8 * hf;
#pragma unroll
  for (int nt = 0; nt < 16; ++nt) {
    int n = nh * 256 + nt * 16 + ml;
    float bv = bias[n];
    float* op = out + (size_t)mg * DD + n;
#pragma unroll
    for (int r = 0; r < 8; ++r) {
      float v = acc2[nt][r] + bv;
      __builtin_nontemporal_store(v > 0.f ? v : 0.f, op + (size_t)r * DD);
    }
  }
}

// ---------------------------------------------------------------------------
extern "C" void kernel_launch(void* const* d_in, const int* in_sizes, int n_in,
                              void* d_out, int out_size, void* d_ws, size_t ws_size,
                              hipStream_t stream) {
  const float* x   = (const float*)d_in[0];  // [8192,512]
  const float* adj = (const float*)d_in[1];  // [8192,8192]
  const float* W   = (const float*)d_in[2];  // [512,512]
  const float* b   = (const float*)d_in[3];  // [512]
  float* out = (float*)d_out;                // [8192,512]

  unsigned short* xT = (unsigned short*)d_ws;                       // 8 MiB
  unsigned short* Wb = (unsigned short*)((char*)d_ws + 8388608);    // 512 KiB

  prep_xT<<<dim3(NN / 32, DD / 32), dim3(32, 8), 0, stream>>>(x, xT);
  prep_W<<<512, 256, 0, stream>>>(W, Wb);
  gnn_fused<<<NN / 64, 256, 0, stream>>>(adj, xT, Wb, b, out);
}
